// mdlstm_9783935500660
// MI455X (gfx1250) — matrix operand order repaired and checked by reading, not compiled, not run on hardware
//
#include <hip/hip_runtime.h>
#include <hip/hip_bf16.h>

// ---------------------------------------------------------------------------
// MD-LSTM (2D LSTM) over 64x64 grid, B=32, D=256, Hd=256.
// Per-pixel gates = [32,768]x[768,1024] GEMM (x | h_up | h_left vs
// Wih | Whh0 | Whh1) via v_wmma_f32_16x16x32_bf16 with f32 accumulate.
// Anti-diagonal wavefront: 127 stream-ordered kernel launches.
// Each wave: one (m-half, gate) pair x TWO 16-chan N-tiles (A-frag reuse),
// with fragment arrays so load clauses issue ahead of the WMMA chain.
// ---------------------------------------------------------------------------

#define IMG 64
#define P_TOT (IMG * IMG)   // 4096 pixels
#define BATCH 32
#define DIN 256
#define HD 256
#define G4 1024             // 4*HD

typedef __attribute__((ext_vector_type(16))) __bf16 v16bf;
typedef __attribute__((ext_vector_type(8)))  float  v8f;
typedef __attribute__((ext_vector_type(8)))  unsigned int v8u;

// ---- fragment loads (CDNA5 WMMA 16-bit layouts, ISA 7.12.2) ----------------
// A 16x32 (MxK): lane<16 -> M=lane, K pairs {0..7} (V0-3), {16..23} (V4-7);
// lane>=16 -> M=lane-16, K pairs {8..15} and {24..31}.
__device__ __forceinline__ v16bf ld_fragA(const __hip_bfloat16* rowbase,
                                          int kstep, int lane) {
    const unsigned int* pd = (const unsigned int*)rowbase + kstep * 16;
    const int s4 = (lane < 16) ? 0 : 4;
    v8u u;
    u[0] = pd[s4 + 0]; u[1] = pd[s4 + 1]; u[2] = pd[s4 + 2]; u[3] = pd[s4 + 3];
    u[4] = pd[8 + s4 + 0]; u[5] = pd[8 + s4 + 1];
    u[6] = pd[8 + s4 + 2]; u[7] = pd[8 + s4 + 3];
    return __builtin_bit_cast(v16bf, u);
}

// B 32x16 (KxN): lane -> N=lane&15; lane<16 holds K {0..7},{16..23}, lane>=16 K {8..15},{24..31}.
// rowbase = weight row n (K-contiguous), since B[k][n] = W[n][k].
__device__ __forceinline__ v16bf ld_fragB(const __hip_bfloat16* rowbase,
                                          int kstep, int lane) {
    const unsigned int* pd = (const unsigned int*)rowbase + kstep * 16;
    const int s4 = (lane < 16) ? 0 : 4;
    v8u u;
    u[0] = pd[s4 + 0]; u[1] = pd[s4 + 1]; u[2] = pd[s4 + 2]; u[3] = pd[s4 + 3];
    u[4] = pd[8 + s4 + 0]; u[5] = pd[8 + s4 + 1]; u[6] = pd[8 + s4 + 2]; u[7] = pd[8 + s4 + 3];
    return __builtin_bit_cast(v16bf, u);
}

// K=256 contribution to TWO accumulators sharing the same A fragments.
// Load order keeps b0 dead before b1 lives (peak ~150 VGPRs) while the
// unrolled arrays let load clauses run ahead of the WMMA chains.
__device__ __forceinline__ void mma_dual_k256(v8f& acc0, v8f& acc1,
                                              const __hip_bfloat16* arow,
                                              const __hip_bfloat16* brow0,
                                              const __hip_bfloat16* brow1,
                                              int lane) {
    v16bf a[8];
#pragma unroll
    for (int ks = 0; ks < 8; ++ks) a[ks] = ld_fragA(arow, ks, lane);
    {
        v16bf b[8];
#pragma unroll
        for (int ks = 0; ks < 8; ++ks) b[ks] = ld_fragB(brow0, ks, lane);
#pragma unroll
        for (int ks = 0; ks < 8; ++ks)
            acc0 = __builtin_amdgcn_wmma_f32_16x16x32_bf16(
                false, a[ks], false, b[ks], (short)0, acc0, false, false);
    }
    {
        v16bf b[8];
#pragma unroll
        for (int ks = 0; ks < 8; ++ks) b[ks] = ld_fragB(brow1, ks, lane);
#pragma unroll
        for (int ks = 0; ks < 8; ++ks)
            acc1 = __builtin_amdgcn_wmma_f32_16x16x32_bf16(
                false, a[ks], false, b[ks], (short)0, acc1, false, false);
    }
}

__device__ __forceinline__ float sigf(float x) {
    return 1.0f / (1.0f + __expf(-x));
}

// ---- one-time conversions --------------------------------------------------
__global__ void k_cvt_weights(const float* __restrict__ wih,
                              const float* __restrict__ whh,
                              __hip_bfloat16* __restrict__ wih_b,
                              __hip_bfloat16* __restrict__ whh_b) {
    int i = blockIdx.x * 256 + threadIdx.x;
    if (i < G4 * DIN)    wih_b[i] = __float2bfloat16(wih[i]);
    if (i < 2 * G4 * HD) whh_b[i] = __float2bfloat16(whh[i]);
}

// input f32 [B][D][P] -> bf16 [P][B][D] (K-contiguous A rows per pixel)
__global__ void k_cvt_input(const float* __restrict__ x,
                            __hip_bfloat16* __restrict__ xbf) {
    long long i = (long long)blockIdx.x * 256 + threadIdx.x;  // < P*B*D
    int d = (int)(i & (DIN - 1));
    int b = (int)((i >> 8) & (BATCH - 1));
    long long p = i >> 13;
    xbf[i] = __float2bfloat16(x[((long long)b * DIN + d) * P_TOT + p]);
}

// ---- wavefront cell kernel: one anti-diagonal ------------------------------
// grid.x = #pixels on diagonal, grid.y = 8 c-tiles (32 hidden chans each).
// 256 threads = 8 waves; wave w: gate q = w>>1, m-half = w&1; each wave
// computes two 16x16 N-tiles (channels cbase..cbase+31) reusing A.
__global__ __launch_bounds__(256)
void k_cell(const __hip_bfloat16* __restrict__ xbf,
            const __hip_bfloat16* __restrict__ wih_b,
            const __hip_bfloat16* __restrict__ whh_b,
            const float* __restrict__ bias_ih,
            const float* __restrict__ bias_hh,
            float* __restrict__ sbuf,             // f32 [P][B][HD]
            __hip_bfloat16* __restrict__ hbf,     // bf16 [P][B][HD]
            float* __restrict__ out,              // f32 [B][HD][P]
            int diag, int rmin) {
    const int r   = rmin + blockIdx.x;
    const int col = diag - r;
    const int p   = r * IMG + col;
    const int cbase = blockIdx.y * 32;

    const int tid  = threadIdx.x;
    const int lane = tid & 31;
    const int wv   = tid >> 5;
    const int q    = wv >> 1;   // gate: 0=i 1=f 2=g 3=o
    const int mh   = wv & 1;    // batch half

    const int m_row = mh * 16 + (lane & 15);           // A row (batch index)
    const int nrow0 = q * 256 + cbase + (lane & 15);   // B rows (gate index)
    const int nrow1 = nrow0 + 16;

    v8f acc0 = {}, acc1 = {};
    // input contribution (always)
    mma_dual_k256(acc0, acc1,
                  xbf + ((size_t)p * BATCH + m_row) * DIN,
                  wih_b + (size_t)nrow0 * DIN,
                  wih_b + (size_t)nrow1 * DIN, lane);
    // up neighbor: h(r-1, col) @ Whh[0]^T
    if (r > 0)
        mma_dual_k256(acc0, acc1,
                      hbf + ((size_t)(p - IMG) * BATCH + m_row) * HD,
                      whh_b + (size_t)nrow0 * HD,
                      whh_b + (size_t)nrow1 * HD, lane);
    // left neighbor: h(r, col-1) @ Whh[1]^T
    if (col > 0)
        mma_dual_k256(acc0, acc1,
                      hbf + ((size_t)(p - 1) * BATCH + m_row) * HD,
                      whh_b + (size_t)(G4 + nrow0) * HD,
                      whh_b + (size_t)(G4 + nrow1) * HD, lane);

    // regroup gates through LDS: [gate][batch 0..31][c_local 0..31]
    __shared__ float g_lds[4][BATCH][32];
    const int cl = lane & 15;
#pragma unroll
    for (int v = 0; v < 8; ++v) {
        int m = mh * 16 + v + ((lane < 16) ? 0 : 8);  // C/D layout (7.12.2)
        g_lds[q][m][cl]      = acc0[v];
        g_lds[q][m][cl + 16] = acc1[v];
    }
    __syncthreads();

    // elementwise LSTM cell: 1024 cells (32 b x 32 c), 4 per thread, f32
    for (int e = tid; e < BATCH * 32; e += 256) {
        const int b   = e >> 5;
        const int cl2 = e & 31;
        const int c   = cbase + cl2;

        const float bi = bias_ih[c]       + bias_hh[c]       + bias_hh[G4 + c];
        const float bf = bias_ih[256 + c] + bias_hh[256 + c] + bias_hh[G4 + 256 + c];
        const float bg = bias_ih[512 + c] + bias_hh[512 + c] + bias_hh[G4 + 512 + c];
        const float bo = bias_ih[768 + c] + bias_hh[768 + c] + bias_hh[G4 + 768 + c];

        const float ig = sigf(g_lds[0][b][cl2] + bi);
        const float fg = sigf(g_lds[1][b][cl2] + bf);
        const float gg = tanhf(g_lds[2][b][cl2] + bg);
        const float og = sigf(g_lds[3][b][cl2] + bo);

        const float s_up   = (r > 0)
            ? sbuf[((size_t)(p - IMG) * BATCH + b) * HD + c] : 0.0f;
        const float s_left = (col > 0)
            ? sbuf[((size_t)(p - 1) * BATCH + b) * HD + c] : 0.0f;

        // reference reuses the same forget gate for both directions
        const float s = fg * (s_up + s_left) + ig * gg;
        const float h = og * tanhf(s);

        sbuf[((size_t)p * BATCH + b) * HD + c] = s;
        hbf [((size_t)p * BATCH + b) * HD + c] = __float2bfloat16(h);
        out [((size_t)b * HD + c) * P_TOT + p] = h;
    }
}

// ---------------------------------------------------------------------------
extern "C" void kernel_launch(void* const* d_in, const int* in_sizes, int n_in,
                              void* d_out, int out_size, void* d_ws, size_t ws_size,
                              hipStream_t stream) {
    const float* input     = (const float*)d_in[0];  // [32][256][4096]
    const float* weight_ih = (const float*)d_in[1];  // [1024][256]
    const float* weight_hh = (const float*)d_in[2];  // [2][1024][256]
    const float* bias_ih   = (const float*)d_in[3];  // [1024]
    const float* bias_hh   = (const float*)d_in[4];  // [2][1024]
    float* out = (float*)d_out;                      // [32][256][4096]

    // workspace carve-up (all region sizes are multiples of 512B)
    __hip_bfloat16* wih_b = (__hip_bfloat16*)d_ws;
    __hip_bfloat16* whh_b = wih_b + (size_t)G4 * DIN;            // +512 KB
    __hip_bfloat16* xbf   = whh_b + (size_t)2 * G4 * HD;         // +1 MB
    __hip_bfloat16* hbf   = xbf   + (size_t)P_TOT * BATCH * DIN; // +64 MB
    float*          sbuf  = (float*)(hbf + (size_t)P_TOT * BATCH * HD); // +64 MB

    // 1) one-time bf16 conversions
    k_cvt_weights<<<(2 * G4 * HD + 255) / 256, 256, 0, stream>>>(
        weight_ih, weight_hh, wih_b, whh_b);
    {
        long long n = (long long)P_TOT * BATCH * DIN;
        k_cvt_input<<<(unsigned)(n / 256), 256, 0, stream>>>(input, xbf);
    }

    // 2) anti-diagonal wavefront: stream order carries the dependency
    for (int d = 0; d <= 2 * (IMG - 1); ++d) {
        int rmin = (d > IMG - 1) ? (d - (IMG - 1)) : 0;
        int rmax = (d < IMG - 1) ? d : (IMG - 1);
        int npix = rmax - rmin + 1;
        dim3 grid(npix, HD / 32);  // pixels x 8 c-tiles
        k_cell<<<grid, 256, 0, stream>>>(xbf, wih_b, whh_b, bias_ih, bias_hh,
                                         sbuf, hbf, out, d, rmin);
    }
}
